// HGCNDecoder_73246372266174
// MI455X (gfx1250) — matrix operand order repaired and checked by reading, not compiled, not run on hardware
//
#include <hip/hip_runtime.h>
#include <hip/hip_bf16.h>
#include <math.h>

typedef __attribute__((ext_vector_type(16))) __bf16 bf16x16;
typedef __attribute__((ext_vector_type(8)))  __bf16 bf16x8;
typedef __attribute__((ext_vector_type(8)))  float  f32x8;

#define MIN_NORM 1e-15f
#define MAXNORM  0.996f   /* (1 - 4e-3)/sqrt(c), c=1 */

__device__ __forceinline__ unsigned short f2bf(float f) {
    unsigned int u = __float_as_uint(f);
    unsigned int r = u + 0x7FFFu + ((u >> 16) & 1u);
    return (unsigned short)(r >> 16);
}

__device__ __forceinline__ float artanh_f(float x) {
    x = fminf(fmaxf(x, -1.0f + 1e-7f), 1.0f - 1e-7f);
    return 0.5f * (log1pf(x) - log1pf(-x));
}

// reduce across a 16-lane half-wave (stays within the WMMA C/D lane halves)
__device__ __forceinline__ float hred16(float v) {
    v += __shfl_xor(v, 1, 32);
    v += __shfl_xor(v, 2, 32);
    v += __shfl_xor(v, 4, 32);
    v += __shfl_xor(v, 8, 32);
    return v;
}

__device__ __forceinline__ bf16x16 mk16(bf16x8 lo, bf16x8 hi) {
    return __builtin_shufflevector(lo, hi, 0,1,2,3,4,5,6,7,8,9,10,11,12,13,14,15);
}

// ---------------------------------------------------------------- prep W: f32 -> bf16
__global__ void k_prepw(const float* __restrict__ W, unsigned short* __restrict__ Wb, int n) {
    int i = blockIdx.x * blockDim.x + threadIdx.x;
    if (i < n) Wb[i] = f2bf(W[i]);
}

// ---------------------------------------------------------------- hyp bias (1 block, 128 thr)
__global__ void k_bias(const float* __restrict__ bias, float* __restrict__ hb,
                       float* __restrict__ hb2) {
    __shared__ float red[4];
    const int tid = threadIdx.x;
    float b = bias[tid];
    float s = b * b;
    s += __shfl_xor(s, 1, 32); s += __shfl_xor(s, 2, 32); s += __shfl_xor(s, 4, 32);
    s += __shfl_xor(s, 8, 32); s += __shfl_xor(s, 16, 32);
    if ((tid & 31) == 0) red[tid >> 5] = s;
    __syncthreads();
    s = red[0] + red[1] + red[2] + red[3];
    float bn = fmaxf(sqrtf(s), MIN_NORM);
    float t  = tanhf(bn);
    float f  = t / bn;                              // expmap0 scale
    float rn = fmaxf(fabsf(t), MIN_NORM);
    float sc = fminf(1.0f, MAXNORM / rn);           // proj scale
    float hv = b * f * sc;
    hb[tid] = hv;
    __syncthreads();
    float s2 = hv * hv;
    s2 += __shfl_xor(s2, 1, 32); s2 += __shfl_xor(s2, 2, 32); s2 += __shfl_xor(s2, 4, 32);
    s2 += __shfl_xor(s2, 8, 32); s2 += __shfl_xor(s2, 16, 32);
    if ((tid & 31) == 0) red[tid >> 5] = s2;
    __syncthreads();
    if (tid == 0) hb2[0] = red[0] + red[1] + red[2] + red[3];
}

// ---------------------------------------------------------------- prep x: bf16 + row norm
__global__ __launch_bounds__(256) void k_prepx(const float* __restrict__ x,
                                               unsigned short* __restrict__ xb,
                                               float* __restrict__ xnorm, int N) {
    __shared__ float red[8];
    const int row = blockIdx.x;
    const int tid = threadIdx.x;
    float v = x[(size_t)row * 256 + tid];
    xb[(size_t)row * 256 + tid] = f2bf(v);
    float s = v * v;
    s += __shfl_xor(s, 1, 32); s += __shfl_xor(s, 2, 32); s += __shfl_xor(s, 4, 32);
    s += __shfl_xor(s, 8, 32); s += __shfl_xor(s, 16, 32);
    if ((tid & 31) == 0) red[tid >> 5] = s;
    __syncthreads();
    if (tid == 0) {
        float t = 0.f;
        #pragma unroll
        for (int i = 0; i < 8; ++i) t += red[i];
        xnorm[row] = fmaxf(sqrtf(t), MIN_NORM);
    }
}

// ---------------------------------------------------------------- HypLinear GEMM + epilogue
// D = x[16x256] @ W.T[256x128] per wave via v_wmma_f32_16x16x32_bf16
__global__ __launch_bounds__(256) void k_hyplinear(const unsigned short* __restrict__ xb,
                                                   const unsigned short* __restrict__ Wb,
                                                   const float* __restrict__ xnorm,
                                                   const float* __restrict__ hb,
                                                   const float* __restrict__ hb2p,
                                                   float* __restrict__ xt, int N) {
    __shared__ __align__(16) unsigned short ldsW[128 * 256];   // 64 KB: full W in bf16
    const int tid = threadIdx.x;

    // cooperative stage of W into LDS (b128 copies)
    {
        const uint4* g = (const uint4*)Wb;
        uint4* s = (uint4*)ldsW;
        #pragma unroll
        for (int i = 0; i < 16; ++i) s[tid + 256 * i] = g[tid + 256 * i];
    }
    __syncthreads();

    const int wave = tid >> 5;
    const int lane = tid & 31;
    const int half = lane >> 4;     // K-half selector per ISA A/B layouts
    const int l16  = lane & 15;

    const int row0 = blockIdx.x * 128 + wave * 16;
    int arow = row0 + l16;
    if (arow >= N) arow = N - 1;    // clamp loads; stores guarded below

    f32x8 acc[8];
    #pragma unroll
    for (int nt = 0; nt < 8; ++nt)
        #pragma unroll
        for (int r = 0; r < 8; ++r) acc[nt][r] = 0.0f;

    const __bf16* xrow = (const __bf16*)xb + (size_t)arow * 256;
    const __bf16* wlds = (const __bf16*)ldsW;

    #pragma unroll
    for (int kt = 0; kt < 8; ++kt) {
        // A fragment 16x32 bf16: lane half selects K sub-rows {0..7,16..23} / {8..15,24..31}
        const int ka = kt * 32 + half * 8;
        bf16x8 alo = *(const bf16x8*)(xrow + ka);
        bf16x8 ahi = *(const bf16x8*)(xrow + ka + 16);
        bf16x16 afrag = mk16(alo, ahi);
        #pragma unroll
        for (int nt = 0; nt < 8; ++nt) {
            // B fragment 32x16: lane = column n, K sub-rows {0..7,16..23} / {8..15,24..31}
            const __bf16* wp = wlds + (size_t)(nt * 16 + l16) * 256 + kt * 32 + half * 8;
            bf16x8 blo = *(const bf16x8*)(wp);
            bf16x8 bhi = *(const bf16x8*)(wp + 16);
            bf16x16 bfrag = mk16(blo, bhi);
            acc[nt] = __builtin_amdgcn_wmma_f32_16x16x32_bf16(
                false, afrag, false, bfrag, (short)0, acc[nt], false, false);
        }
    }

    // ---------- fused hyperbolic epilogue (c = 1) ----------
    float hbc[8];
    #pragma unroll
    for (int nt = 0; nt < 8; ++nt) hbc[nt] = hb[nt * 16 + l16];
    const float y2 = hb2p[0];

    #pragma unroll
    for (int r = 0; r < 8; ++r) {
        const int row = row0 + r + 8 * half;   // C/D layout: lanes 16-31 hold M = r+8
        const bool valid = row < N;

        // mx_norm over 128 cols of this row
        float s = 0.f;
        #pragma unroll
        for (int nt = 0; nt < 8; ++nt) { float m = acc[nt][r]; s += m * m; }
        s = hred16(s);
        float mxn = fmaxf(sqrtf(s), MIN_NORM);
        float xn  = xnorm[valid ? row : 0];

        // mobius_matvec tail: res = tanh(mxn/xn * artanh(xn)) * mx / mxn
        float t  = tanhf(mxn / xn * artanh_f(xn));
        float f  = t / mxn;
        float rn = fmaxf(fabsf(t), MIN_NORM);
        f *= fminf(1.0f, MAXNORM / rn);        // proj
        float hn = fminf(rn, MAXNORM);         // ||h||

        // mobius_add(h, hyp_bias)
        float xy = 0.f;
        #pragma unroll
        for (int nt = 0; nt < 8; ++nt) xy += acc[nt][r] * f * hbc[nt];
        xy = hred16(xy);
        float x2  = hn * hn;
        float ca  = 1.0f + 2.0f * xy + y2;
        float cb  = 1.0f - x2;
        float den = fmaxf(1.0f + 2.0f * xy + x2 * y2, MIN_NORM);
        float inv = 1.0f / den;

        float g[8]; float gs = 0.f;
        #pragma unroll
        for (int nt = 0; nt < 8; ++nt) {
            g[nt] = (ca * acc[nt][r] * f + cb * hbc[nt]) * inv;
            gs += g[nt] * g[nt];
        }
        gs = hred16(gs);
        float gn  = fmaxf(sqrtf(gs), MIN_NORM);
        float ps  = fminf(1.0f, MAXNORM / gn); // proj
        float gnc = fminf(gn, MAXNORM);
        // logmap0: xt = artanh(gnc)/gnc * g_projected
        float lf = artanh_f(gnc) / gnc * ps;

        if (valid) {
            float* orow = xt + (size_t)row * 128 + l16;
            #pragma unroll
            for (int nt = 0; nt < 8; ++nt) orow[nt * 16] = g[nt] * lf;
        }
    }
}

// ---------------------------------------------------------------- SpMM scatter: 1 wave / edge
__global__ __launch_bounds__(256) void k_spmm(const int* __restrict__ ei,
                                              const float* __restrict__ ew,
                                              const float* __restrict__ xt,
                                              float* __restrict__ out, int E) {
    const int e = blockIdx.x * 8 + (threadIdx.x >> 5);
    if (e >= E) return;
    const int lane = threadIdx.x & 31;
    const int dst = ei[e];          // edge_index[0] = segment (row)
    const int src = ei[E + e];      // edge_index[1] = gathered source (col)
    const float w = ew[e];
    const float4 m = ((const float4*)(xt + (size_t)src * 128))[lane];
    float* op = out + (size_t)dst * 128 + lane * 4;
    atomicAdd(op + 0, m.x * w);
    atomicAdd(op + 1, m.y * w);
    atomicAdd(op + 2, m.z * w);
    atomicAdd(op + 3, m.w * w);
}

// ---------------------------------------------------------------- final expmap0/proj chain
__global__ __launch_bounds__(128) void k_final(float* __restrict__ out, int N) {
    __shared__ float red[4];
    const int row = blockIdx.x;
    const int tid = threadIdx.x;
    float g = out[(size_t)row * 128 + tid];
    float s = g * g;
    s += __shfl_xor(s, 1, 32); s += __shfl_xor(s, 2, 32); s += __shfl_xor(s, 4, 32);
    s += __shfl_xor(s, 8, 32); s += __shfl_xor(s, 16, 32);
    if ((tid & 31) == 0) red[tid >> 5] = s;
    __syncthreads();
    s = red[0] + red[1] + red[2] + red[3];
    float n1 = fmaxf(sqrtf(s), MIN_NORM);
    // h = proj(expmap0(agg))
    float t1  = tanhf(n1);
    float f1  = t1 / n1;
    float rn1 = fmaxf(fabsf(t1), MIN_NORM);
    float sc1 = fminf(1.0f, MAXNORM / rn1);
    float hn  = fminf(rn1, MAXNORM);
    // proj(expmap0(logmap0(h)))
    float l   = artanh_f(hn);
    float t2  = tanhf(l);
    float f2  = t2 / fmaxf(hn, MIN_NORM);
    float rn2 = fmaxf(fabsf(t2), MIN_NORM);
    float sc2 = fminf(1.0f, MAXNORM / rn2);
    out[(size_t)row * 128 + tid] = g * f1 * sc1 * f2 * sc2;
}

extern "C" void kernel_launch(void* const* d_in, const int* in_sizes, int n_in,
                              void* d_out, int out_size, void* d_ws, size_t ws_size,
                              hipStream_t stream) {
    const float* x    = (const float*)d_in[0];   // [N,256]
    const int*   ei   = (const int*)d_in[1];     // [2,E]
    const float* ew   = (const float*)d_in[2];   // [E]
    const float* W    = (const float*)d_in[3];   // [128,256]
    const float* bias = (const float*)d_in[4];   // [128]

    const int D = 256, C = 128;
    const int N = in_sizes[0] / D;
    const int E = in_sizes[2];

    char* ws = (char*)d_ws;
    size_t off = 0;
    unsigned short* xb = (unsigned short*)(ws + off); off += (size_t)N * D * 2;  // x bf16
    unsigned short* Wb = (unsigned short*)(ws + off); off += (size_t)C * D * 2;  // W bf16
    float* hb    = (float*)(ws + off); off += (size_t)C * 4;                     // hyp bias
    float* hb2   = (float*)(ws + off); off += 256;                               // ||hb||^2
    float* xnorm = (float*)(ws + off); off += (((size_t)N * 4) + 255) & ~255ull; // row norms
    float* xt    = (float*)(ws + off); off += (size_t)N * C * 4;                 // tangent feats

    float* out = (float*)d_out;

    hipMemsetAsync(out, 0, (size_t)N * C * sizeof(float), stream);   // agg accumulator
    k_prepw<<<(C * D + 255) / 256, 256, 0, stream>>>(W, Wb, C * D);
    k_bias<<<1, 128, 0, stream>>>(bias, hb, hb2);
    k_prepx<<<N, 256, 0, stream>>>(x, xb, xnorm, N);
    k_hyplinear<<<(N + 127) / 128, 256, 0, stream>>>(xb, Wb, xnorm, hb, hb2, xt, N);
    k_spmm<<<(E + 7) / 8, 256, 0, stream>>>(ei, ew, xt, out, E);
    k_final<<<N, 128, 0, stream>>>(out, N);
}
